// Structure_Decoder_6141803233545
// MI455X (gfx1250) — matrix operand order repaired and checked by reading, not compiled, not run on hardware
//
#include <hip/hip_runtime.h>

#define NHID 128

typedef __attribute__((ext_vector_type(16))) __bf16 bfrag16; // A/B operand: 8 VGPRs
typedef __attribute__((ext_vector_type(8)))  float  accf8;   // C/D accumulator: 8 VGPRs

union FragU { bfrag16 f; uint4 q[2]; };

__device__ inline unsigned short f2bf(float x) {
    union { float f; unsigned u; } c; c.f = x;
    unsigned u = c.u;
    unsigned r = (u + 0x7FFFu + ((u >> 16) & 1u)) >> 16; // round-to-nearest-even
    return (unsigned short)r;
}
__device__ inline float bf2f(unsigned short h) {
    union { unsigned u; float f; } c; c.u = ((unsigned)h) << 16;
    return c.f;
}

// ---------------- init: deg = 1 (self loop), agg = 0 ----------------
__global__ void k_init(float* __restrict__ deg, float* __restrict__ agg, int N) {
    int i = blockIdx.x * blockDim.x + threadIdx.x;
    if (i < N) deg[i] = 1.0f;
    if (i < N * NHID) agg[i] = 0.0f;
}

// ---------------- degree: count in-edges ----------------
__global__ void k_deg(const int* __restrict__ ei, float* __restrict__ deg, int E) {
    int e = blockIdx.x * blockDim.x + threadIdx.x;
    if (e < E) atomicAdd(&deg[ei[E + e]], 1.0f);
}

// ---------------- dinv = rsqrt(deg)  (deg >= 1 always) ----------------
__global__ void k_dinv(float* __restrict__ deg, int N) {
    int i = blockIdx.x * blockDim.x + threadIdx.x;
    if (i < N) deg[i] = rsqrtf(deg[i]);
}

// ---------------- h0 = x @ W  (one row per block, 128 threads) ----------------
__global__ __launch_bounds__(NHID) void k_xw(const float* __restrict__ x,
                                             const float* __restrict__ W,
                                             float* __restrict__ h0, int N) {
    __shared__ float xs[NHID];
    int row = blockIdx.x;
    int t = threadIdx.x;
    xs[t] = x[(size_t)row * NHID + t];
    __syncthreads();
    float acc = 0.0f;
#pragma unroll 8
    for (int k = 0; k < NHID; ++k)
        acc = fmaf(xs[k], W[k * NHID + t], acc);
    h0[(size_t)row * NHID + t] = acc;
}

// ---------------- scatter-add: one wave per edge (incl. self loops) ----------------
__global__ void k_scatter(const int* __restrict__ ei, const float* __restrict__ h0,
                          const float* __restrict__ dinv, float* __restrict__ agg,
                          int E, int N) {
    int wid  = (blockIdx.x * blockDim.x + threadIdx.x) >> 5;
    int lane = threadIdx.x & 31;
    if (wid >= E + N) return;
    int s, d;
    if (wid < E) { s = ei[wid]; d = ei[E + wid]; }
    else         { s = wid - E; d = s; }
    float nrm = dinv[s] * dinv[d];
    float4 v = ((const float4*)(h0 + (size_t)s * NHID))[lane];
    float* a = agg + (size_t)d * NHID + lane * 4;
    atomicAdd(a + 0, v.x * nrm);
    atomicAdd(a + 1, v.y * nrm);
    atomicAdd(a + 2, v.z * nrm);
    atomicAdd(a + 3, v.w * nrm);
}

// ---------------- relu(+b) and bf16 hi/lo split ----------------
__global__ void k_fin(const float* __restrict__ agg, const float* __restrict__ b,
                      unsigned short* __restrict__ hhi, unsigned short* __restrict__ hlo,
                      int N) {
    int i = blockIdx.x * blockDim.x + threadIdx.x;
    if (i >= N * NHID) return;
    float v = agg[i] + b[i & (NHID - 1)];
    v = fmaxf(v, 0.0f);
    unsigned short hi = f2bf(v);
    unsigned short lo = f2bf(v - bf2f(hi));
    hhi[i] = hi;
    hlo[i] = lo;
}

// ---------------- S = h @ h^T via v_wmma_f32_16x16x32_bf16, hi/lo split ----------------
// WG = 256 threads = 8 waves arranged 2 (row) x 4 (col); each wave computes a 64x64
// output block (4x4 tiles of 16x16) -> WG covers 128x256. K = 128 = 4 steps of 32.
// Per K-step: 48 WMMA vs 32 b128 loads (ratio 1.5); 192 WMMA per wave total.
__global__ __launch_bounds__(256) void k_gemm(const unsigned short* __restrict__ hhi,
                                              const unsigned short* __restrict__ hlo,
                                              float* __restrict__ out, int N) {
    const int wave = threadIdx.x >> 5;
    const int lane = threadIdx.x & 31;
    const int wrow = wave >> 2;   // 0..1
    const int wcol = wave & 3;    // 0..3
    const int hi   = lane >> 4;   // lane group
    const int l15  = lane & 15;

    const int r0 = blockIdx.y * 128 + wrow * 64; // wave's output row base (4 tiles)
    const int c0 = blockIdx.x * 256 + wcol * 64; // wave's output col base (4 tiles)

    accf8 acc[4][4] = {};

    // per-lane source rows for A (rows) and B (cols, since B = h^T) fragments
    int arow[4], brow[4];
#pragma unroll
    for (int t = 0; t < 4; ++t) {
        int r = r0 + 16 * t + l15; arow[t] = r < N ? r : N - 1;
        int c = c0 + 16 * t + l15; brow[t] = c < N ? c : N - 1;
    }

#pragma unroll
    for (int ks = 0; ks < 4; ++ks) {
        const int akb = ks * 32 + hi * 8;   // A: lanes 0-15 -> K{0..7,16..23}, 16-31 -> +8
        const int bkb = ks * 32 + hi * 8;   // B: lanes 0-15 -> K{0..7,16..23}, 16-31 -> +8

        FragU Ah[4], Al[4], Bh[4], Bl[4];
#pragma unroll
        for (int t = 0; t < 4; ++t) {
            const uint4* ph = (const uint4*)(hhi + (size_t)arow[t] * NHID + akb);
            const uint4* pl = (const uint4*)(hlo + (size_t)arow[t] * NHID + akb);
            Ah[t].q[0] = ph[0]; Ah[t].q[1] = ph[2]; // +16 elems = +2 uint4
            Al[t].q[0] = pl[0]; Al[t].q[1] = pl[2];
        }
#pragma unroll
        for (int u = 0; u < 4; ++u) {
            const uint4* ph = (const uint4*)(hhi + (size_t)brow[u] * NHID + bkb);
            const uint4* pl = (const uint4*)(hlo + (size_t)brow[u] * NHID + bkb);
            Bh[u].q[0] = ph[0]; Bh[u].q[1] = ph[2]; // +16 elems = +2 uint4
            Bl[u].q[0] = pl[0]; Bl[u].q[1] = pl[2];
        }

#pragma unroll
        for (int t = 0; t < 4; ++t)
#pragma unroll
            for (int u = 0; u < 4; ++u) {
                acc[t][u] = __builtin_amdgcn_wmma_f32_16x16x32_bf16(
                    false, Ah[t].f, false, Bh[u].f, (short)0, acc[t][u], false, false);
                acc[t][u] = __builtin_amdgcn_wmma_f32_16x16x32_bf16(
                    false, Ah[t].f, false, Bl[u].f, (short)0, acc[t][u], false, false);
                acc[t][u] = __builtin_amdgcn_wmma_f32_16x16x32_bf16(
                    false, Al[t].f, false, Bh[u].f, (short)0, acc[t][u], false, false);
            }
    }

    // C/D layout: VGPR r -> M = r + hi*8, N = lane&15 (64B contiguous per half-wave).
    // Output is written exactly once and never re-read on-chip -> non-temporal stores
    // keep the 192MB L2 reserved for the hot h operands.
#pragma unroll
    for (int t = 0; t < 4; ++t)
#pragma unroll
        for (int u = 0; u < 4; ++u) {
            int col = c0 + 16 * u + l15;
            if (col >= N) continue;
#pragma unroll
            for (int r = 0; r < 8; ++r) {
                int row = r0 + 16 * t + hi * 8 + r;
                if (row < N)
                    __builtin_nontemporal_store(acc[t][u][r], &out[(size_t)row * N + col]);
            }
        }
}

extern "C" void kernel_launch(void* const* d_in, const int* in_sizes, int n_in,
                              void* d_out, int out_size, void* d_ws, size_t ws_size,
                              hipStream_t stream) {
    const float* x  = (const float*)d_in[0];
    const int*   ei = (const int*)d_in[1];
    const float* W  = (const float*)d_in[2];
    const float* b  = (const float*)d_in[3];
    float* out = (float*)d_out;

    const int N = in_sizes[0] / NHID;   // 10000
    const int E = in_sizes[1] / 2;      // 320000

    // workspace carve-up (all 16B aligned): ~15.4 MB total
    float* deg = (float*)d_ws;                       // N      (becomes dinv)
    float* h0  = deg + N;                            // N*128
    float* agg = h0 + (size_t)N * NHID;              // N*128
    unsigned short* hhi = (unsigned short*)(agg + (size_t)N * NHID); // N*128 bf16
    unsigned short* hlo = hhi + (size_t)N * NHID;                    // N*128 bf16

    k_init<<<(N * NHID + 255) / 256, 256, 0, stream>>>(deg, agg, N);
    k_xw<<<N, NHID, 0, stream>>>(x, W, h0, N);
    k_deg<<<(E + 255) / 256, 256, 0, stream>>>(ei, deg, E);
    k_dinv<<<(N + 255) / 256, 256, 0, stream>>>(deg, N);

    long long scatter_threads = (long long)(E + N) * 32;
    k_scatter<<<(unsigned)((scatter_threads + 255) / 256), 256, 0, stream>>>(
        ei, h0, deg, agg, E, N);

    k_fin<<<(N * NHID + 255) / 256, 256, 0, stream>>>(agg, b, hhi, hlo, N);

    dim3 grid((N + 255) / 256, (N + 127) / 128);
    k_gemm<<<grid, 256, 0, stream>>>(hhi, hlo, out, N);
}
